// PinSAGE_29618094473883
// MI455X (gfx1250) — matrix operand order repaired and checked by reading, not compiled, not run on hardware
//
#include <hip/hip_runtime.h>
#include <hip/hip_bf16.h>

typedef __bf16 v16bf __attribute__((ext_vector_type(16)));
typedef float  v8f   __attribute__((ext_vector_type(8)));

#define N_NODES 50000
#define N_PAD   50016      // padded row count (multiple of 32) for GEMM A inputs
#define N_EDGES 800000
#define IN_CH   128
#define HID_CH  256
#define OUT_CH  128

// ---------------------------------------------------------------- utilities
__device__ __forceinline__ unsigned f32_to_bf16_rne(float f) {
    unsigned u = __float_as_uint(f);
    unsigned r = u + 0x7FFFu + ((u >> 16) & 1u);   // round-to-nearest-even
    return r >> 16;
}
__device__ __forceinline__ unsigned pack2_bf16(float lo, float hi) {
    return f32_to_bf16_rne(lo) | (f32_to_bf16_rne(hi) << 16);
}
__device__ __forceinline__ void atomic_add_f32(float* p, float v) {
    __hip_atomic_fetch_add(p, v, __ATOMIC_RELAXED, __HIP_MEMORY_SCOPE_AGENT);
}

// ---------------------------------------------------------------- zero init
__global__ void k_zero(float* __restrict__ p, long long n) {
    long long i = (long long)blockIdx.x * blockDim.x + threadIdx.x;
    long long s = (long long)gridDim.x * blockDim.x;
    for (; i < n; i += s) p[i] = 0.0f;
}

// ------------------------------------------------- f32 -> bf16 row-major copy
__global__ void k_cvt_bf16(const float* __restrict__ src, unsigned* __restrict__ dst,
                           long long n4) {
    long long i = (long long)blockIdx.x * blockDim.x + threadIdx.x;
    if (i >= n4) return;
    float4 f = ((const float4*)src)[i];
    uint2 o;
    o.x = pack2_bf16(f.x, f.y);
    o.y = pack2_bf16(f.z, f.w);
    ((uint2*)dst)[i] = o;
}

// ------------------------------------------- pack f32 weights -> WMMA B-frag
// dword index = ((kt*nT + nt)*32 + lane)*8 + i
//   lane 0-15  : K = kt*32 +     16*(i>>2) + 2(i&3),+1 ; N = nt*16 + lane
//   lane 16-31 : K = kt*32 + 8 + 16*(i>>2) + 2(i&3),+1 ; N = nt*16 + (lane-16)
__global__ void k_pack_w(const float* __restrict__ W, unsigned* __restrict__ P,
                         int Kin, int Nout) {
    int total = (Kin * Nout) >> 1;
    int idx = blockIdx.x * blockDim.x + threadIdx.x;
    if (idx >= total) return;
    int i     = idx & 7;
    int lane  = (idx >> 3) & 31;
    int tile  = idx >> 8;
    int nT    = Nout >> 4;
    int nt    = tile % nT;
    int kt    = tile / nT;
    int n     = nt * 16 + (lane & 15);
    int k     = kt * 32 + ((lane >> 4) << 3) + ((i >> 2) << 4) + ((i & 3) << 1);
    P[idx] = pack2_bf16(W[k * Nout + n], W[(k + 1) * Nout + n]);
}

// ---------------------------------------------------------------- degrees
__global__ void k_deg(const int* __restrict__ ei, float* __restrict__ deg) {
    int e = blockIdx.x * blockDim.x + threadIdx.x;
    if (e >= N_EDGES) return;
    atomic_add_f32(&deg[ei[N_EDGES + e]], 1.0f);
}

// ------------------------------------------------ scatter-add of messages
// one thread handles 4 channels of one edge; C/4 threads per edge
__global__ void k_scatter(const int* __restrict__ ei, const float* __restrict__ X,
                          float* __restrict__ agg, int C, int gshift) {
    int epb = blockDim.x >> gshift;                       // edges per block
    int e   = blockIdx.x * epb + (threadIdx.x >> gshift);
    if (e >= N_EDGES) return;
    int c4  = (threadIdx.x & ((1 << gshift) - 1)) << 2;
    int src = ei[e];
    int dst = ei[N_EDGES + e];
    const float4 v = *(const float4*)&X[(long long)src * C + c4];
    float* a = &agg[(long long)dst * C + c4];
    atomic_add_f32(a + 0, v.x);
    atomic_add_f32(a + 1, v.y);
    atomic_add_f32(a + 2, v.z);
    atomic_add_f32(a + 3, v.w);
}

// ------------------------------------- mean + f32->bf16 pack, fused
__global__ void k_mean_cvt(const float* __restrict__ agg, const float* __restrict__ deg,
                           unsigned* __restrict__ dst, long long n4, int cshift) {
    long long i = (long long)blockIdx.x * blockDim.x + threadIdx.x;
    if (i >= n4) return;
    long long base = i << 2;
    float inv = 1.0f / fmaxf(deg[base >> cshift], 1.0f);
    float4 f = ((const float4*)agg)[i];
    uint2 o;
    o.x = pack2_bf16(f.x * inv, f.y * inv);
    o.y = pack2_bf16(f.z * inv, f.w * inv);
    ((uint2*)dst)[i] = o;
}

// ------------------------------------------------ A fragment (pre-packed bf16)
// 16-bit A 16x32 layout: lane m (0-15): v0..3 = K k0..k0+7 ; v4..7 = K k0+16..k0+23
// with k0 = kt*32 + 8*(lane>=16). Row-major bf16 => two b128 loads.
__device__ __forceinline__ v16bf load_a_frag(const unsigned short* __restrict__ rowp,
                                             int k0) {
    union { v16bf v; uint4 q[2]; } r;
    r.q[0] = *(const uint4*)(rowp + k0);
    r.q[1] = *(const uint4*)(rowp + k0 + 16);
    return r.v;
}

// ------------------------------------- fused dual GEMM: Out = A1@W1 + A2@W2 + b
// per wave: 32(M) x 64(N) tile (2 row sub-tiles share B frags); 8 waves/block.
// A inputs are padded to N_PAD rows; stores guarded to N_NODES.
template <int KIN, int NOUT, bool RELU>
__global__ void __launch_bounds__(256)
k_gemm_dual(const unsigned short* __restrict__ A1, const unsigned short* __restrict__ A2,
            const unsigned* __restrict__ P1, const unsigned* __restrict__ P2,
            const float* __restrict__ bias, float* __restrict__ Out) {
    constexpr int NT  = NOUT >> 4;   // 16-wide N tiles in full matrix
    constexpr int NKT = KIN >> 5;    // K steps of 32
    constexpr int RT  = N_PAD / 32;  // 32-row tiles

    const int lane    = threadIdx.x & 31;
    const int wave    = threadIdx.x >> 5;
    const int rowTile = blockIdx.x * 8 + wave;
    if (rowTile >= RT) return;                    // wave-uniform exit
    const int colStrip = blockIdx.y;              // 64-wide strip
    const int khalf    = lane >> 4;
    const int m        = lane & 15;

    const unsigned short* a1row0 = A1 + ((long long)rowTile * 32 + m) * KIN;
    const unsigned short* a2row0 = A2 + ((long long)rowTile * 32 + m) * KIN;
    const unsigned short* a1row1 = a1row0 + (long long)16 * KIN;
    const unsigned short* a2row1 = a2row0 + (long long)16 * KIN;

    v8f acc[2][4];
#pragma unroll
    for (int s = 0; s < 2; ++s)
#pragma unroll
        for (int t = 0; t < 4; ++t)
#pragma unroll
            for (int r = 0; r < 8; ++r) acc[s][t][r] = 0.0f;

#pragma unroll
    for (int kt = 0; kt < NKT; ++kt) {
        const int k0 = kt * 32 + khalf * 8;

        // ---- batch all loads for this K step ----
        union { v16bf v; uint4 q[2]; } b1[4], b2[4];
#pragma unroll
        for (int t = 0; t < 4; ++t) {
            const int nt = colStrip * 4 + t;
            const long long boff = (((long long)kt * NT + nt) * 32 + lane) * 8;
            b1[t].q[0] = *(const uint4*)&P1[boff];
            b1[t].q[1] = *(const uint4*)&P1[boff + 4];
            b2[t].q[0] = *(const uint4*)&P2[boff];
            b2[t].q[1] = *(const uint4*)&P2[boff + 4];
        }
        v16bf a1s0 = load_a_frag(a1row0, k0);
        v16bf a2s0 = load_a_frag(a2row0, k0);
        v16bf a1s1 = load_a_frag(a1row1, k0);
        v16bf a2s1 = load_a_frag(a2row1, k0);

        // ---- 16 WMMAs, B fragments reused across both row sub-tiles ----
#pragma unroll
        for (int t = 0; t < 4; ++t) {
            acc[0][t] = __builtin_amdgcn_wmma_f32_16x16x32_bf16(
                false, a1s0, false, b1[t].v, (short)0, acc[0][t], false, false);
            acc[0][t] = __builtin_amdgcn_wmma_f32_16x16x32_bf16(
                false, a2s0, false, b2[t].v, (short)0, acc[0][t], false, false);
            acc[1][t] = __builtin_amdgcn_wmma_f32_16x16x32_bf16(
                false, a1s1, false, b1[t].v, (short)0, acc[1][t], false, false);
            acc[1][t] = __builtin_amdgcn_wmma_f32_16x16x32_bf16(
                false, a2s1, false, b2[t].v, (short)0, acc[1][t], false, false);
        }
    }

    // C/D layout: lane 0-15: N=lane, VGPR r -> M=r ; lane 16-31: M=8+r
    const int colBase = colStrip * 64;
#pragma unroll
    for (int s = 0; s < 2; ++s) {
#pragma unroll
        for (int t = 0; t < 4; ++t) {
            const int col = colBase + t * 16 + m;
            const float bv = bias[col];
#pragma unroll
            for (int r = 0; r < 8; ++r) {
                const long long row = (long long)rowTile * 32 + s * 16 + khalf * 8 + r;
                float v = acc[s][t][r] + bv;
                if (RELU) v = fmaxf(v, 0.0f);
                if (row < N_NODES) Out[row * NOUT + col] = v;
            }
        }
    }
}

// ---------------------------------------------------------- row log-softmax
__global__ void k_logsoftmax(float* __restrict__ Z) {
    __shared__ float red[OUT_CH];
    const int row = blockIdx.x;
    const int c   = threadIdx.x;
    float v = Z[(long long)row * OUT_CH + c];
    red[c] = v;
    __syncthreads();
#pragma unroll
    for (int s = OUT_CH / 2; s > 0; s >>= 1) {
        if (c < s) red[c] = fmaxf(red[c], red[c + s]);
        __syncthreads();
    }
    const float mx = red[0];
    __syncthreads();
    const float e = expf(v - mx);
    red[c] = e;
    __syncthreads();
#pragma unroll
    for (int s = OUT_CH / 2; s > 0; s >>= 1) {
        if (c < s) red[c] += red[c + s];
        __syncthreads();
    }
    const float lse = logf(red[0]) + mx;
    Z[(long long)row * OUT_CH + c] = v - lse;
}

// ---------------------------------------------------------------- launcher
extern "C" void kernel_launch(void* const* d_in, const int* in_sizes, int n_in,
                              void* d_out, int out_size, void* d_ws, size_t ws_size,
                              hipStream_t stream) {
    const float* x   = (const float*)d_in[0];
    const int*   ei  = (const int*)d_in[1];   // edge_index [2, E]
    const float* W1l = (const float*)d_in[2];
    const float* W1r = (const float*)d_in[3];
    const float* b1  = (const float*)d_in[4];
    const float* W2l = (const float*)d_in[5];
    const float* W2r = (const float*)d_in[6];
    const float* b2  = (const float*)d_in[7];
    float* out = (float*)d_out;

    char* ws = (char*)d_ws;
    size_t off = 0;
    auto carve = [&](size_t bytes) -> void* {
        void* p = ws + off;
        off = (off + bytes + 255) & ~(size_t)255;
        return p;
    };
    float* deg  = (float*)carve((size_t)N_NODES * 4);
    float* agg1 = (float*)carve((size_t)N_NODES * IN_CH * 4);     // zeroed
    float* agg2 = (float*)carve((size_t)N_NODES * HID_CH * 4);    // zeroed
    float* h    = (float*)carve((size_t)N_NODES * HID_CH * 4);    // GEMM1 out (f32)
    // bf16 GEMM inputs, padded to N_PAD rows (pad rows read as garbage, stores guarded)
    unsigned short* xb  = (unsigned short*)carve((size_t)N_PAD * IN_CH * 2);
    unsigned short* a1b = (unsigned short*)carve((size_t)N_PAD * IN_CH * 2);
    unsigned short* hb  = (unsigned short*)carve((size_t)N_PAD * HID_CH * 2);
    unsigned short* a2b = (unsigned short*)carve((size_t)N_PAD * HID_CH * 2);
    unsigned* pW1l = (unsigned*)carve((size_t)IN_CH * HID_CH * 2);
    unsigned* pW1r = (unsigned*)carve((size_t)IN_CH * HID_CH * 2);
    unsigned* pW2l = (unsigned*)carve((size_t)HID_CH * OUT_CH * 2);
    unsigned* pW2r = (unsigned*)carve((size_t)HID_CH * OUT_CH * 2);

    // zero deg + agg1 + agg2 (contiguous region from deg to start of h)
    long long nz = (long long)(h - deg);
    k_zero<<<2048, 256, 0, stream>>>(deg, nz);

    // pack weights into WMMA bf16 B-fragment layout
    k_pack_w<<<(IN_CH * HID_CH / 2 + 255) / 256, 256, 0, stream>>>(W1l, pW1l, IN_CH, HID_CH);
    k_pack_w<<<(IN_CH * HID_CH / 2 + 255) / 256, 256, 0, stream>>>(W1r, pW1r, IN_CH, HID_CH);
    k_pack_w<<<(HID_CH * OUT_CH / 2 + 255) / 256, 256, 0, stream>>>(W2l, pW2l, HID_CH, OUT_CH);
    k_pack_w<<<(HID_CH * OUT_CH / 2 + 255) / 256, 256, 0, stream>>>(W2r, pW2r, HID_CH, OUT_CH);

    // degrees
    k_deg<<<(N_EDGES + 255) / 256, 256, 0, stream>>>(ei, deg);

    // x -> bf16 (GEMM1 root input)
    {
        long long n4 = (long long)N_NODES * IN_CH / 4;
        k_cvt_bf16<<<(int)((n4 + 255) / 256), 256, 0, stream>>>(x, (unsigned*)xb, n4);
    }

    // ---- layer 1: mean aggregate x -> agg1 -> a1b (bf16)
    k_scatter<<<N_EDGES * (IN_CH / 4) / 256, 256, 0, stream>>>(ei, x, agg1, IN_CH, 5);
    {
        long long n4 = (long long)N_NODES * IN_CH / 4;
        k_mean_cvt<<<(int)((n4 + 255) / 256), 256, 0, stream>>>(agg1, deg, (unsigned*)a1b, n4, 7);
    }

    // h = relu(agg1 @ W1_l + x @ W1_r + b1)   [50000 x 256]
    {
        dim3 grid((N_PAD / 32 + 7) / 8, HID_CH / 64);
        k_gemm_dual<IN_CH, HID_CH, true><<<grid, 256, 0, stream>>>(
            a1b, xb, pW1l, pW1r, b1, h);
    }

    // h -> bf16 (GEMM2 root input)
    {
        long long n4 = (long long)N_NODES * HID_CH / 4;
        k_cvt_bf16<<<(int)((n4 + 255) / 256), 256, 0, stream>>>(h, (unsigned*)hb, n4);
    }

    // ---- layer 2: mean aggregate h -> agg2 -> a2b (bf16)
    k_scatter<<<N_EDGES * (HID_CH / 4) / 256, 256, 0, stream>>>(ei, h, agg2, HID_CH, 6);
    {
        long long n4 = (long long)N_NODES * HID_CH / 4;
        k_mean_cvt<<<(int)((n4 + 255) / 256), 256, 0, stream>>>(agg2, deg, (unsigned*)a2b, n4, 8);
    }

    // out = agg2 @ W2_l + h @ W2_r + b2   [50000 x 128]
    {
        dim3 grid((N_PAD / 32 + 7) / 8, OUT_CH / 64);
        k_gemm_dual<HID_CH, OUT_CH, false><<<grid, 256, 0, stream>>>(
            a2b, hb, pW2l, pW2r, b2, out);
    }

    // in-place row-wise log-softmax
    k_logsoftmax<<<N_NODES, OUT_CH, 0, stream>>>(out);
}
